// SDConv_5669356830988
// MI455X (gfx1250) — matrix operand order repaired and checked by reading, not compiled, not run on hardware
//
#include <hip/hip_runtime.h>

typedef __attribute__((ext_vector_type(16))) _Float16 v16h;
typedef __attribute__((ext_vector_type(8)))  _Float16 v8h;
typedef __attribute__((ext_vector_type(8)))  float    v8f;
typedef __attribute__((ext_vector_type(4)))  float    v4f;

#define Bn   4
#define Ln   2048
#define Dn   768
#define Hn   12
#define Kn   9
#define IPGn 64
#define PADn 4

// ------------------------------------------------------------------
// Kernel A: v[m][n] = sum_k q[m][k] * pt_w[n][k] + pt_b[n]
//   M = B*L = 8192, N = 768, K = 768.  WG tile 64(M) x 128(N), Kblk=32.
//   8 waves; wave computes 16x64 (4 accumulator tiles).
// ------------------------------------------------------------------
#define AM 64
#define AN 128
#define AK 32

__global__ __launch_bounds__(256) void gemm_v_kernel(
    const float* __restrict__ q, const float* __restrict__ ptw,
    const float* __restrict__ ptb, float* __restrict__ v)
{
  __shared__ alignas(32) _Float16 Alds[AM * AK];   // [m][k], row stride 32
  __shared__ alignas(32) _Float16 Blds[AN * AK];   // [n][k], row stride 32

  const int tid  = threadIdx.x;
  const int wv   = tid >> 5;
  const int lane = tid & 31;
  const int half = lane >> 4;
  const int m0   = (blockIdx.x & 127) * AM;        // 128 M-blocks
  const int n0   = (blockIdx.x >> 7) * AN;         // 6 N-blocks
  const int mw   = (wv & 3) * 16;                  // wave M sub-tile
  const int nwv  = (wv >> 2) * 64;                 // wave N strip
  const int mrow = mw + (lane & 15);

  v8f acc[4];
#pragma unroll
  for (int t = 0; t < 4; ++t)
#pragma unroll
    for (int i = 0; i < 8; ++i) acc[t][i] = 0.0f;

  for (int k0 = 0; k0 < Dn; k0 += AK) {
    // A tile: 64x32 fp32 -> f16 LDS (512 float4, 2 per thread)
#pragma unroll
    for (int i = 0; i < 2; ++i) {
      int f = tid * 2 + i;
      int r = f >> 3, c = (f & 7) * 4;
      v4f d = *(const v4f*)(q + (size_t)(m0 + r) * Dn + k0 + c);
      _Float16* dst = &Alds[r * AK + c];
      dst[0] = (_Float16)d[0]; dst[1] = (_Float16)d[1];
      dst[2] = (_Float16)d[2]; dst[3] = (_Float16)d[3];
    }
    // B tile: pt_w rows n0..n0+127, cols k0..k0+31 (k-contiguous = B-frag order)
#pragma unroll
    for (int i = 0; i < 4; ++i) {
      int f = tid + i * 256;
      int r = f >> 3, c = (f & 7) * 4;
      v4f d = *(const v4f*)(ptw + (size_t)(n0 + r) * Dn + k0 + c);
      _Float16* dst = &Blds[r * AK + c];
      dst[0] = (_Float16)d[0]; dst[1] = (_Float16)d[1];
      dst[2] = (_Float16)d[2]; dst[3] = (_Float16)d[3];
    }
    __syncthreads();

    // A fragment: lane holds row mrow, 16 contiguous K values offset by 16*half
    v16h a = *(const v16h*)&Alds[mrow * AK + 16 * half];

#pragma unroll
    for (int nt = 0; nt < 4; ++nt) {
      int n = nwv + nt * 16 + (lane & 15);
      // B fragment: 16 contiguous K values of column n, offset by 16*half
      v16h bf = *(const v16h*)&Blds[n * AK + 16 * half];
      acc[nt] = __builtin_amdgcn_wmma_f32_16x16x32_f16(
          false, a, false, bf, (short)0, acc[nt], false, false);
    }
    __syncthreads();
  }

#pragma unroll
  for (int nt = 0; nt < 4; ++nt) {
    int col = n0 + nwv + nt * 16 + (lane & 15);
    float bias = ptb[col];
#pragma unroll
    for (int j = 0; j < 8; ++j) {
      int row = m0 + mw + j + 8 * half;           // C layout: M = j + 8*half
      v[(size_t)row * Dn + col] = acc[nt][j] + bias;
    }
  }
}

// ------------------------------------------------------------------
// Kernel B: fused depthwise conv + grouped pointwise (WMMA) + attn
//           kernel + softmax + 9-tap window-weighted sum of v.
//   One WG handles 16 consecutive l positions of one batch.
// ------------------------------------------------------------------
__global__ __launch_bounds__(256) void sdconv_kernel(
    const float* __restrict__ q,    const float* __restrict__ dww,
    const float* __restrict__ pww_g,const float* __restrict__ pwb,
    const float* __restrict__ akw,  const float* __restrict__ akb,
    const float* __restrict__ v,    float* __restrict__ out)
{
  __shared__ alignas(32) _Float16 pww[Hn * IPGn * IPGn];  // 98304 B  [h][o][i]
  __shared__ alignas(32) _Float16 dwb[16 * Dn];           // 24576 B  [l][d]
  __shared__ alignas(32) _Float16 cab[16 * Dn];           // 24576 B  [l][d]
  __shared__ float wb[16 * Hn * Kn];                      //  6912 B  [l][h][k]

  const int tid  = threadIdx.x;
  const int wv   = tid >> 5;
  const int lane = tid & 31;
  const int half = lane >> 4;
  const int b    = blockIdx.x >> 7;            // 128 tiles per batch
  const int l0   = (blockIdx.x & 127) * 16;
  const size_t qbase = (size_t)b * Ln * Dn;

  // Phase 0: pw_w fp32 -> f16 LDS ([h][o][i] flat layout kept as-is)
  for (int i = tid; i < Hn * IPGn * IPGn; i += 256)
    pww[i] = (_Float16)pww_g[i];

  // Phase 1: depthwise conv  dw[l][d] = sum_k q[l+k-4][d] * dw_w[d][k]
  for (int idx = tid; idx < 16 * Dn; idx += 256) {
    int l = idx / Dn, d = idx - l * Dn;
    float s = 0.0f;
#pragma unroll
    for (int k = 0; k < Kn; ++k) {
      int lq = l0 + l + k - PADn;
      if (lq >= 0 && lq < Ln)
        s += q[qbase + (size_t)lq * Dn + d] * dww[d * Kn + k];
    }
    dwb[idx] = (_Float16)s;
  }
  __syncthreads();

  // Phase 2: per-head pointwise pw[l][o] = sum_i dw[l][h*64+i]*pw_w[h][o][i]
  //          then ca[l][d] = (pw + pw_b[d]) * q[l][d]   (WMMA, 48 jobs / 8 waves)
  for (int job = wv; job < Hn * 4; job += 8) {
    int h = job >> 2, nt = job & 3;
    int m = lane & 15;
    v16h a[2];
#pragma unroll
    for (int s = 0; s < 2; ++s) {
      int base = m * Dn + h * IPGn + s * 32;
      a[s] = *(const v16h*)&dwb[base + 16 * half];
    }
    int o = nt * 16 + (lane & 15);
    v8f c;
#pragma unroll
    for (int i = 0; i < 8; ++i) c[i] = 0.0f;
#pragma unroll
    for (int s = 0; s < 2; ++s) {
      v16h bf = *(const v16h*)&pww[h * 4096 + o * IPGn + s * 32 + 16 * half];
      c = __builtin_amdgcn_wmma_f32_16x16x32_f16(
          false, a[s], false, bf, (short)0, c, false, false);
    }
    int d = h * IPGn + o;
    float bias = pwb[d];
#pragma unroll
    for (int j = 0; j < 8; ++j) {
      int mloc = j + 8 * half;
      float ca = (c[j] + bias) * q[qbase + (size_t)(l0 + mloc) * Dn + d];
      cab[mloc * Dn + d] = (_Float16)ca;
    }
  }
  __syncthreads();

  // Phase 3: kern[l][h][k] = sum_i ca[l][h*64+i]*ak_w[h][k][i] + ak_b ; softmax_k
  if (tid < 16 * Hn) {
    int l = tid & 15, h = tid >> 4;
    float kern[Kn];
#pragma unroll
    for (int k = 0; k < Kn; ++k) {
      float s = akb[h * Kn + k];
      const float* w = &akw[(h * Kn + k) * IPGn];
      const _Float16* c = &cab[l * Dn + h * IPGn];
#pragma unroll
      for (int i = 0; i < IPGn; ++i) s += (float)c[i] * w[i];
      kern[k] = s;
    }
    float mx = kern[0];
#pragma unroll
    for (int k = 1; k < Kn; ++k) mx = fmaxf(mx, kern[k]);
    float se = 0.0f;
#pragma unroll
    for (int k = 0; k < Kn; ++k) { kern[k] = __expf(kern[k] - mx); se += kern[k]; }
    float inv = 1.0f / se;
#pragma unroll
    for (int k = 0; k < Kn; ++k) wb[(l * Hn + h) * Kn + k] = kern[k] * inv;
  }
  __syncthreads();

  // Phase 4: out[l][d] = sum_k v[l+k-4][d] * w[l][h][k]
  for (int idx = tid; idx < 16 * Dn; idx += 256) {
    int m = idx / Dn, d = idx - m * Dn;
    int h = d >> 6;
    const float* wl = &wb[(m * Hn + h) * Kn];
    float acc = 0.0f;
#pragma unroll
    for (int k = 0; k < Kn; ++k) {
      int lv = l0 + m + k - PADn;
      if (lv >= 0 && lv < Ln)
        acc += v[qbase + (size_t)lv * Dn + d] * wl[k];
    }
    out[qbase + (size_t)(l0 + m) * Dn + d] = acc;
  }
}

extern "C" void kernel_launch(void* const* d_in, const int* in_sizes, int n_in,
                              void* d_out, int out_size, void* d_ws, size_t ws_size,
                              hipStream_t stream) {
  const float* q   = (const float*)d_in[0];
  const float* dww = (const float*)d_in[1];
  const float* pww = (const float*)d_in[2];
  const float* pwb = (const float*)d_in[3];
  const float* akw = (const float*)d_in[4];
  const float* akb = (const float*)d_in[5];
  const float* ptw = (const float*)d_in[6];
  const float* ptb = (const float*)d_in[7];
  float* out = (float*)d_out;
  float* v   = (float*)d_ws;   // B*L*D fp32 = 25.2 MB scratch

  // v = query @ pt_w^T + pt_b  (M-blocks=128, N-blocks=6)
  gemm_v_kernel<<<dim3(128 * 6), 256, 0, stream>>>(q, ptw, ptb, v);
  // fused SDConv over 16-position tiles
  sdconv_kernel<<<dim3(Bn * Ln / 16), 256, 0, stream>>>(q, dww, pww, pwb, akw, akb, v, out);
}
